// APPNPNet_74251394613578
// MI455X (gfx1250) — compile-verified, matrix operand order checked by reading, not run on hardware
//
#include <hip/hip_runtime.h>
#include <hip/hip_bf16.h>

typedef __attribute__((ext_vector_type(16))) _Float16 v16h;
typedef __attribute__((ext_vector_type(8)))  float    v8f;

#define N_FEAT 512
#define HID    128
#define NCLS   3
#define KPROP  10
#define ALPHA  0.1f
#define EPSB   1e-5f

// LDS row strides (padded to dodge bank conflicts, keep 16B alignment)
#define W1_LD 520   // halves: 512 + 8
#define H1_LD 136   // halves: 128 + 8
#define HS_LD 132   // floats: 128 + 4
#define SMEM_BYTES (HID * W1_LD * 2)   // 133120 B, the max of all phases
                                       // (2 blocks/WGP fit in the 320KB pool)

union Frag16 {
    v16h v;
    _Float16 h[16];
    uint4 q[2];
};

// ---------------------------------------------------------------------------
// Fused MLP: h0 = (relu(bn1(x@w1^T+b1)) + relu(bn2(.@w2^T+b2))) @ w3^T + b3
// One block = 128 nodes. 8 waves; wave w owns rows [16w,16w+16) x 128 cols.
// ---------------------------------------------------------------------------
__global__ __launch_bounds__(256) void mlp_wmma_kernel(
    const float* __restrict__ x,
    const float* __restrict__ w1, const float* __restrict__ b1,
    const float* __restrict__ g1, const float* __restrict__ be1,
    const float* __restrict__ m1, const float* __restrict__ v1,
    const float* __restrict__ w2, const float* __restrict__ b2,
    const float* __restrict__ g2, const float* __restrict__ be2,
    const float* __restrict__ m2, const float* __restrict__ v2,
    const float* __restrict__ w3, const float* __restrict__ b3,
    float* __restrict__ h0out, int nnodes)
{
    extern __shared__ char smem[];
    _Float16* w1s = (_Float16*)smem;                      // phase 1: [128][W1_LD]
    _Float16* h1s = (_Float16*)smem;                      // phase 2: [128][H1_LD]
    _Float16* w2s = (_Float16*)(smem + HID * H1_LD * 2);  // phase 2: [128][H1_LD]
    float*    hs  = (float*)smem;                         // phase 3: [128][HS_LD]

    const int tid   = threadIdx.x;
    const int wave  = tid >> 5;
    const int lane  = tid & 31;
    const int half_ = lane >> 4;   // K-half selector in 16-bit WMMA layout
    const int r     = lane & 15;   // M (A) / N (B,C,D) row index within tile
    const int node0 = blockIdx.x * 128;

    // ---- stage w1 (f32 -> f16) into LDS ----
    for (int i = tid; i < (HID * N_FEAT) / 4; i += 256) {
        const int n = i >> 7;            // / (512/4)
        const int k = (i & 127) << 2;
        const float4 f = *(const float4*)(w1 + n * N_FEAT + k);
        _Float16* d = w1s + n * W1_LD + k;
        d[0] = (_Float16)f.x; d[1] = (_Float16)f.y;
        d[2] = (_Float16)f.z; d[3] = (_Float16)f.w;
    }
    __syncthreads();

    // ---- GEMM1: [16 x 512] @ w1^T -> 8 tiles of [16 x 16] ----
    v8f acc[8];
#pragma unroll
    for (int nt = 0; nt < 8; ++nt) acc[nt] = (v8f)(0.0f);

    int row = node0 + wave * 16 + r;
    if (row >= nnodes) row = nnodes - 1;          // clamp; stores are guarded
    const float* xr = x + (long)row * N_FEAT;

    // Explicit double-buffer of the raw f32 A-tile: issue loads one K-tile
    // ahead so a full nt-loop (8 WMMAs + 16 ds_load_b128) covers the latency.
    float4 nf0, nf1, nf2, nf3;
    {
        const int k0 = 8 * (2 * 0 + half_);
        const int k1 = 8 * (2 * 1 + half_);
        nf0 = *(const float4*)(xr + k0);
        nf1 = *(const float4*)(xr + k0 + 4);
        nf2 = *(const float4*)(xr + k1);
        nf3 = *(const float4*)(xr + k1 + 4);
    }

    for (int kt = 0; kt < N_FEAT / 32; ++kt) {
        Frag16 a;
        a.h[0]  = (_Float16)nf0.x; a.h[1]  = (_Float16)nf0.y;
        a.h[2]  = (_Float16)nf0.z; a.h[3]  = (_Float16)nf0.w;
        a.h[4]  = (_Float16)nf1.x; a.h[5]  = (_Float16)nf1.y;
        a.h[6]  = (_Float16)nf1.z; a.h[7]  = (_Float16)nf1.w;
        a.h[8]  = (_Float16)nf2.x; a.h[9]  = (_Float16)nf2.y;
        a.h[10] = (_Float16)nf2.z; a.h[11] = (_Float16)nf2.w;
        a.h[12] = (_Float16)nf3.x; a.h[13] = (_Float16)nf3.y;
        a.h[14] = (_Float16)nf3.z; a.h[15] = (_Float16)nf3.w;

        if (kt + 1 < N_FEAT / 32) {
            const int kb = (kt + 1) * 32;
            const int k0 = kb + 8 * (2 * 0 + half_);
            const int k1 = kb + 8 * (2 * 1 + half_);
            nf0 = *(const float4*)(xr + k0);
            nf1 = *(const float4*)(xr + k0 + 4);
            nf2 = *(const float4*)(xr + k1);
            nf3 = *(const float4*)(xr + k1 + 4);
        }
        if (kt + 2 < N_FEAT / 32)   // WGP-scope speculative prefetch, 2 tiles out
            __builtin_prefetch(xr + (kt + 2) * 32, 0, 3);

#pragma unroll
        for (int nt = 0; nt < 8; ++nt) {
            Frag16 b;
            const int n = nt * 16 + r;
#pragma unroll
            for (int g = 0; g < 2; ++g) {
                const int k = kt * 32 + 8 * (2 * g + half_);
                b.q[g] = *(const uint4*)(w1s + n * W1_LD + k);  // 16B LDS read
            }
            acc[nt] = __builtin_amdgcn_wmma_f32_16x16x32_f16(
                false, a.v, false, b.v, (short)0, acc[nt], false, false);
        }
    }

    // ---- BN1 + ReLU in registers (per-column scale/shift) ----
#pragma unroll
    for (int nt = 0; nt < 8; ++nt) {
        const int c = nt * 16 + r;
        const float s = g1[c] * rsqrtf(v1[c] + EPSB);
        const float t = (b1[c] - m1[c]) * s + be1[c];
#pragma unroll
        for (int j = 0; j < 8; ++j) {
            const float vv = acc[nt][j] * s + t;
            acc[nt][j] = vv > 0.0f ? vv : 0.0f;   // acc now holds h1 (f32)
        }
    }
    __syncthreads();   // all waves done reading w1s; reuse LDS

    // ---- store h1 (f16) + stage w2 (f16) ----
#pragma unroll
    for (int nt = 0; nt < 8; ++nt) {
        const int c = nt * 16 + r;
#pragma unroll
        for (int j = 0; j < 8; ++j) {
            const int mrow = wave * 16 + j + 8 * half_;
            h1s[mrow * H1_LD + c] = (_Float16)acc[nt][j];
        }
    }
    for (int i = tid; i < (HID * HID) / 4; i += 256) {
        const int n = i >> 5;
        const int k = (i & 31) << 2;
        const float4 f = *(const float4*)(w2 + n * HID + k);
        _Float16* d = w2s + n * H1_LD + k;
        d[0] = (_Float16)f.x; d[1] = (_Float16)f.y;
        d[2] = (_Float16)f.z; d[3] = (_Float16)f.w;
    }
    __syncthreads();

    // ---- GEMM2: [16 x 128] @ w2^T ----
    v8f acc2[8];
#pragma unroll
    for (int nt = 0; nt < 8; ++nt) acc2[nt] = (v8f)(0.0f);

#pragma unroll
    for (int kt = 0; kt < HID / 32; ++kt) {
        Frag16 a;
        const int arow = wave * 16 + r;
#pragma unroll
        for (int g = 0; g < 2; ++g) {
            const int k = kt * 32 + 8 * (2 * g + half_);
            a.q[g] = *(const uint4*)(h1s + arow * H1_LD + k);
        }
#pragma unroll
        for (int nt = 0; nt < 8; ++nt) {
            Frag16 b;
            const int n = nt * 16 + r;
#pragma unroll
            for (int g = 0; g < 2; ++g) {
                const int k = kt * 32 + 8 * (2 * g + half_);
                b.q[g] = *(const uint4*)(w2s + n * H1_LD + k);
            }
            acc2[nt] = __builtin_amdgcn_wmma_f32_16x16x32_f16(
                false, a.v, false, b.v, (short)0, acc2[nt], false, false);
        }
    }

    // ---- BN2 + ReLU + residual (h = h1 + relu(bn2(h1@w2^T+b2))) ----
#pragma unroll
    for (int nt = 0; nt < 8; ++nt) {
        const int c = nt * 16 + r;
        const float s = g2[c] * rsqrtf(v2[c] + EPSB);
        const float t = (b2[c] - m2[c]) * s + be2[c];
#pragma unroll
        for (int j = 0; j < 8; ++j) {
            float vv = acc2[nt][j] * s + t;
            vv = vv > 0.0f ? vv : 0.0f;
            acc[nt][j] += vv;                    // residual: h in acc
        }
    }
    __syncthreads();   // done reading h1s / w2s

    // ---- spill h (f32) to LDS for the tiny GEMM3 ----
#pragma unroll
    for (int nt = 0; nt < 8; ++nt) {
        const int c = nt * 16 + r;
#pragma unroll
        for (int j = 0; j < 8; ++j) {
            const int mrow = wave * 16 + j + 8 * half_;
            hs[mrow * HS_LD + c] = acc[nt][j];
        }
    }
    __syncthreads();

    // ---- GEMM3: [128 x 128] @ w3^T[128 x 3] + b3 -> h0 ----
    for (int i = tid; i < 128 * NCLS; i += 256) {
        const int rw = i / NCLS;
        const int c  = i % NCLS;
        const int node = node0 + rw;
        if (node < nnodes) {
            float sum = b3[c];
            const float* wr = w3 + c * HID;
            const float* hr = hs + rw * HS_LD;
#pragma unroll 8
            for (int k = 0; k < HID; ++k) sum += hr[k] * wr[k];
            h0out[(long)node * NCLS + c] = sum;
        }
    }
}

// ---------------------------------------------------------------------------
// APPNP propagation kernels
// ---------------------------------------------------------------------------
__global__ __launch_bounds__(256) void deg_init_kernel(float* __restrict__ deg, int n) {
    const int i = blockIdx.x * 256 + threadIdx.x;
    if (i < n) deg[i] = 1.0f;   // self-loop
}

__global__ __launch_bounds__(256) void deg_edge_kernel(const int* __restrict__ dst,
                                                       float* __restrict__ deg, int e) {
    const int i = blockIdx.x * 256 + threadIdx.x;
    if (i < e) atomicAdd(&deg[dst[i]], 1.0f);
}

__global__ __launch_bounds__(256) void deg_fin_kernel(float* __restrict__ dinv, int n) {
    const int i = blockIdx.x * 256 + threadIdx.x;
    if (i < n) dinv[i] = rsqrtf(dinv[i]);   // deg >= 1 always
}

// zw = ALPHA*h0 + (1-ALPHA)*dinv^2*zr      (self-loop term + teleport)
__global__ __launch_bounds__(256) void prop_init_kernel(
    float* __restrict__ zw, const float* __restrict__ zr,
    const float* __restrict__ h0, const float* __restrict__ dinv, int n)
{
    const int i = blockIdx.x * 256 + threadIdx.x;
    if (i < n) {
        const float d2 = dinv[i] * dinv[i] * (1.0f - ALPHA);
        const long b = (long)i * NCLS;
#pragma unroll
        for (int c = 0; c < NCLS; ++c)
            zw[b + c] = ALPHA * h0[b + c] + d2 * zr[b + c];
    }
}

// zw[dst] += (1-ALPHA)*dinv[src]*dinv[dst]*zr[src]
__global__ __launch_bounds__(256) void prop_edge_kernel(
    float* __restrict__ zw, const float* __restrict__ zr,
    const int* __restrict__ src, const int* __restrict__ dst,
    const float* __restrict__ dinv, int e)
{
    const int i = blockIdx.x * 256 + threadIdx.x;
    if (i < e) {
        const int s = src[i];
        const int d = dst[i];
        const float w = (1.0f - ALPHA) * dinv[s] * dinv[d];
        const long sb = (long)s * NCLS;
        const long db = (long)d * NCLS;
#pragma unroll
        for (int c = 0; c < NCLS; ++c)
            atomicAdd(&zw[db + c], w * zr[sb + c]);
    }
}

__global__ __launch_bounds__(256) void log_softmax_kernel(
    float* __restrict__ out, const float* __restrict__ z, int n)
{
    const int i = blockIdx.x * 256 + threadIdx.x;
    if (i < n) {
        const long b = (long)i * NCLS;
        const float a0 = z[b], a1 = z[b + 1], a2 = z[b + 2];
        const float m = fmaxf(a0, fmaxf(a1, a2));
        const float lse = logf(expf(a0 - m) + expf(a1 - m) + expf(a2 - m));
        out[b]     = a0 - m - lse;
        out[b + 1] = a1 - m - lse;
        out[b + 2] = a2 - m - lse;
    }
}

// ---------------------------------------------------------------------------
extern "C" void kernel_launch(void* const* d_in, const int* in_sizes, int n_in,
                              void* d_out, int out_size, void* d_ws, size_t ws_size,
                              hipStream_t stream) {
    const float* x   = (const float*)d_in[0];
    const int*   ei  = (const int*)d_in[1];
    const float* w1  = (const float*)d_in[2];
    const float* b1  = (const float*)d_in[3];
    const float* g1  = (const float*)d_in[4];
    const float* be1 = (const float*)d_in[5];
    const float* m1  = (const float*)d_in[6];
    const float* v1  = (const float*)d_in[7];
    const float* w2  = (const float*)d_in[8];
    const float* b2  = (const float*)d_in[9];
    const float* g2  = (const float*)d_in[10];
    const float* be2 = (const float*)d_in[11];
    const float* m2  = (const float*)d_in[12];
    const float* v2  = (const float*)d_in[13];
    const float* w3  = (const float*)d_in[14];
    const float* b3  = (const float*)d_in[15];

    const int N = in_sizes[0] / N_FEAT;
    const int E = in_sizes[1] / 2;
    const int* src = ei;
    const int* dst = ei + E;

    float* ws   = (float*)d_ws;
    float* h0   = ws;
    float* zA   = h0 + (size_t)N * NCLS;
    float* zB   = zA + (size_t)N * NCLS;
    float* dinv = zB + (size_t)N * NCLS;

    float* out = (float*)d_out;

    // Fused MLP (WMMA): 782 blocks x 256 threads, 130 KB dynamic LDS
    mlp_wmma_kernel<<<(N + 127) / 128, 256, SMEM_BYTES, stream>>>(
        x, w1, b1, g1, be1, m1, v1, w2, b2, g2, be2, m2, v2, w3, b3, h0, N);

    const int nb = (N + 255) / 256;
    const int eb = (E + 255) / 256;
    deg_init_kernel<<<nb, 256, 0, stream>>>(dinv, N);
    deg_edge_kernel<<<eb, 256, 0, stream>>>(dst, dinv, E);
    deg_fin_kernel<<<nb, 256, 0, stream>>>(dinv, N);

    const float* zr = h0;
    float* zw = zB;
    for (int it = 0; it < KPROP; ++it) {
        prop_init_kernel<<<nb, 256, 0, stream>>>(zw, zr, h0, dinv, N);
        prop_edge_kernel<<<eb, 256, 0, stream>>>(zw, zr, src, dst, dinv, E);
        const float* written = zw;
        zw = (it == 0) ? zA : (float*)zr;
        zr = written;
    }
    log_softmax_kernel<<<nb, 256, 0, stream>>>(out, zr, N);
}
